// MultiHeadSelfAttention_9259949490332
// MI455X (gfx1250) — matrix operand order repaired and checked by reading, not compiled, not run on hardware
//
#include <hip/hip_runtime.h>
#include <hip/hip_bf16.h>

typedef _Float16 h16;
typedef __attribute__((ext_vector_type(16))) _Float16 v16h;
typedef __attribute__((ext_vector_type(8)))  _Float16 v8h;
typedef __attribute__((ext_vector_type(8)))  float    v8f;

#define EMBED   1024
#define HEADS   16
#define HD      64
#define SEQ     2048
#define BATCH   4
#define BS      (BATCH * SEQ)     // 8192 tokens
#define NQKV    (3 * EMBED)      // 3072
#define ATT_SCALE 0.125f         // 1/sqrt(64)
#define PSTRIDE 72               // LDS P-tile row stride in halfs (144B, 16B aligned)

// ---------------------------------------------------------------------------
// WMMA helper (wave32, 16x16x32 f16 -> f32)
// ---------------------------------------------------------------------------
static __device__ __forceinline__ v8f wmma_f16(v16h a, v16h b, v8f c) {
    return __builtin_amdgcn_wmma_f32_16x16x32_f16(
        /*neg_a=*/false, a, /*neg_b=*/false, b,
        /*c_mod=*/(short)0, c, /*reuse_a=*/false, /*reuse_b=*/false);
}

// A-fragment (16x32, MxK) from row-major [rows, ld] f16.
// Lanes 0-15: M=lane, VGPR0-3 = K(k0+hi*8..+7), VGPR4-7 = K(k0+16+hi*8..)
static __device__ __forceinline__ v16h frag_rowK(const h16* base, long ld,
                                                 int row0, int k0, int lane) {
    const int l = lane & 15, hi = lane >> 4;
    const h16* p = base + (long)(row0 + l) * ld + k0 + hi * 8;
    v8h lo = *(const v8h*)p;
    v8h hh = *(const v8h*)(p + 16);
    v16h r;
#pragma unroll
    for (int i = 0; i < 8; ++i) { r[i] = lo[i]; r[i + 8] = hh[i]; }
    return r;
}

// B-fragment (32x16, KxN) from K-contiguous [N, K] storage.
// Lanes 0-15: N=n0+lane, VGPR0-3 = K(k0+hi*8..+7), VGPR4-7 = K(k0+16+hi*8..)
static __device__ __forceinline__ v16h frag_colK(const h16* base, long ld,
                                                 int n0, int k0, int lane) {
    const int l = lane & 15, hi = lane >> 4;
    const h16* p = base + (long)(n0 + l) * ld + k0 + hi * 8;
    v8h lo = *(const v8h*)p;
    v8h hh = *(const v8h*)(p + 16);
    v16h r;
#pragma unroll
    for (int i = 0; i < 8; ++i) { r[i] = lo[i]; r[i + 8] = hh[i]; }
    return r;
}

// ---------------------------------------------------------------------------
// Conversion kernels
// ---------------------------------------------------------------------------
__global__ void cvt_f32_to_f16(const float* __restrict__ s, h16* __restrict__ d, int n) {
    int i = blockIdx.x * blockDim.x + threadIdx.x;
    if (i < n) d[i] = (h16)s[i];
}

// s: [rows, cols] f32 -> d: [cols, rows] f16 (transposed)
__global__ void cvt_transpose_f16(const float* __restrict__ s, h16* __restrict__ d,
                                  int rows, int cols) {
    int i = blockIdx.x * blockDim.x + threadIdx.x;
    if (i < rows * cols) {
        int r = i / cols, c = i % cols;
        d[(long)c * rows + r] = (h16)s[i];
    }
}

// ---------------------------------------------------------------------------
// GEMM1: Xh[8192,1024] @ WqkvT[3072,1024] + bias -> Q(scaled)/K/Vt(permuted)
// 8 waves (2x4), wave tile 32x64, K-step 32. Operands stream through 192MB L2.
// V keys are stored pre-permuted within 32-blocks so the attention kernel can
// use packed P stores with a matching K-permutation (P@V is perm-invariant).
// ---------------------------------------------------------------------------
__global__ __launch_bounds__(256) void gemm_qkv(
    const h16* __restrict__ X, const h16* __restrict__ Wt,
    const float* __restrict__ bias,
    h16* __restrict__ q, h16* __restrict__ k, h16* __restrict__ vt) {
    const int lane = threadIdx.x & 31;
    const int wid  = threadIdx.x >> 5;
    const int row0 = blockIdx.x * 64 + (wid & 1) * 32;
    const int n0   = blockIdx.y * 256 + (wid >> 1) * 64;

    v8f C[2][4];
#pragma unroll
    for (int i = 0; i < 2; ++i)
#pragma unroll
        for (int j = 0; j < 4; ++j)
#pragma unroll
            for (int r = 0; r < 8; ++r) C[i][j][r] = 0.0f;

    for (int k0 = 0; k0 < EMBED; k0 += 32) {
        v16h a0 = frag_rowK(X, EMBED, row0,      k0, lane);
        v16h a1 = frag_rowK(X, EMBED, row0 + 16, k0, lane);
#pragma unroll
        for (int tj = 0; tj < 4; ++tj) {
            v16h bf = frag_colK(Wt, EMBED, n0 + tj * 16, k0, lane);
            C[0][tj] = wmma_f16(a0, bf, C[0][tj]);
            C[1][tj] = wmma_f16(a1, bf, C[1][tj]);
        }
    }

    const int l = lane & 15, hi = lane >> 4;
#pragma unroll
    for (int tj = 0; tj < 4; ++tj) {
        const int n     = n0 + tj * 16 + l;
        const float bv  = bias[n];
        const int which = n >> 10;          // 0=Q 1=K 2=V
        const int c     = n & 1023;
        const int hh    = c >> 6;
        const int d     = c & 63;
#pragma unroll
        for (int ti = 0; ti < 2; ++ti) {
#pragma unroll
            for (int r = 0; r < 8; ++r) {
                const int m  = row0 + ti * 16 + hi * 8 + r;
                const int b_ = m >> 11;
                const int s  = m & 2047;
                const int bh = b_ * HEADS + hh;
                const float val = C[ti][tj][r] + bv;
                if (which == 0)
                    q[((long)bh * SEQ + s) * HD + d] = (h16)(val * ATT_SCALE);
                else if (which == 1)
                    k[((long)bh * SEQ + s) * HD + d] = (h16)val;
                else {
                    // permute key index within its 32-block: c -> ((c&15)<<1)|(c>>4)
                    const int sp = (s & ~31) | ((s & 15) << 1) | ((s >> 4) & 1);
                    vt[((long)bh * HD + d) * SEQ + sp] = (h16)val;
                }
            }
        }
    }
}

// ---------------------------------------------------------------------------
// Flash attention: one wave owns 32 query rows of one (b,h); 64-key tiles.
// Score accum layout: VGPR r, lanes 0-15 -> row r, lanes 16-31 -> row 8+r.
// P is written to LDS with interleaved key order so each lane's (col l, col
// l+16) pair packs into one b32 store; V was stored with the same permutation.
// ---------------------------------------------------------------------------
__global__ __launch_bounds__(256) void attn_flash(
    const h16* __restrict__ q, const h16* __restrict__ k,
    const h16* __restrict__ vt, h16* __restrict__ ctx) {
    __shared__ h16 plds[8][32 * PSTRIDE];   // per-wave 32x64 P tile

    const int lane = threadIdx.x & 31;
    const int wid  = threadIdx.x >> 5;
    const int w    = blockIdx.x * 8 + wid;
    const int qt   = w & 63;
    const int bh   = w >> 6;
    const int q0   = qt * 32;
    const int l    = lane & 15, hi = lane >> 4;

    const h16* qb = q  + (long)bh * SEQ * HD;
    const h16* kb = k  + (long)bh * SEQ * HD;
    const h16* vb = vt + (long)bh * HD * SEQ;

    v8f  C[2][4];
    float mrow[2][8], lrow[2][8];
#pragma unroll
    for (int ti = 0; ti < 2; ++ti)
#pragma unroll
        for (int r = 0; r < 8; ++r) { mrow[ti][r] = -3.0e38f; lrow[ti][r] = 0.0f; }
#pragma unroll
    for (int ti = 0; ti < 2; ++ti)
#pragma unroll
        for (int nj = 0; nj < 4; ++nj)
#pragma unroll
            for (int r = 0; r < 8; ++r) C[ti][nj][r] = 0.0f;

    for (int key0 = 0; key0 < SEQ; key0 += 64) {
        // ---- S = Q @ K^T over a 32x64 tile (Q pre-scaled) ----
        v8f S[2][4];
#pragma unroll
        for (int ti = 0; ti < 2; ++ti)
#pragma unroll
            for (int tj = 0; tj < 4; ++tj)
#pragma unroll
                for (int r = 0; r < 8; ++r) S[ti][tj][r] = 0.0f;
#pragma unroll
        for (int kk = 0; kk < 2; ++kk) {
            v16h qa0 = frag_rowK(qb, HD, q0,      kk * 32, lane);   // L0-resident
            v16h qa1 = frag_rowK(qb, HD, q0 + 16, kk * 32, lane);
#pragma unroll
            for (int tj = 0; tj < 4; ++tj) {
                v16h kf = frag_colK(kb, HD, key0 + tj * 16, kk * 32, lane);
                S[0][tj] = wmma_f16(qa0, kf, S[0][tj]);
                S[1][tj] = wmma_f16(qa1, kf, S[1][tj]);
            }
        }

        // ---- online softmax; packed P stores (interleaved key order) ----
#pragma unroll
        for (int ti = 0; ti < 2; ++ti) {
#pragma unroll
            for (int r = 0; r < 8; ++r) {
                float mx = fmaxf(fmaxf(S[ti][0][r], S[ti][1][r]),
                                 fmaxf(S[ti][2][r], S[ti][3][r]));
                mx = fmaxf(mx, __shfl_xor(mx, 1));
                mx = fmaxf(mx, __shfl_xor(mx, 2));
                mx = fmaxf(mx, __shfl_xor(mx, 4));
                mx = fmaxf(mx, __shfl_xor(mx, 8));
                const float mnew = fmaxf(mrow[ti][r], mx);
                const float corr = __expf(mrow[ti][r] - mnew);
                const float p0 = __expf(S[ti][0][r] - mnew);
                const float p1 = __expf(S[ti][1][r] - mnew);
                const float p2 = __expf(S[ti][2][r] - mnew);
                const float p3 = __expf(S[ti][3][r] - mnew);
                float rs = (p0 + p1) + (p2 + p3);
                rs += __shfl_xor(rs, 1);
                rs += __shfl_xor(rs, 2);
                rs += __shfl_xor(rs, 4);
                rs += __shfl_xor(rs, 8);
                lrow[ti][r] = lrow[ti][r] * corr + rs;
                mrow[ti][r] = mnew;
#pragma unroll
                for (int nj = 0; nj < 4; ++nj) C[ti][nj][r] *= corr;
                const int row = ti * 16 + hi * 8 + r;
                union { unsigned int u; h16 h[2]; } pk;
                pk.h[0] = (h16)p0; pk.h[1] = (h16)p1;          // cols l, l+16
                *(unsigned int*)&plds[wid][row * PSTRIDE + 2 * l] = pk.u;
                pk.h[0] = (h16)p2; pk.h[1] = (h16)p3;          // cols 32+l, 48+l
                *(unsigned int*)&plds[wid][row * PSTRIDE + 32 + 2 * l] = pk.u;
            }
        }

        // ---- C += P @ V (K-dim permutation matches on both operands) ----
        // Same-wave DS ops are in-order on CDNA5: no barrier needed.
#pragma unroll
        for (int kc = 0; kc < 2; ++kc) {
            v16h pa0 = frag_rowK(&plds[wid][0], PSTRIDE, 0,  kc * 32, lane);
            v16h pa1 = frag_rowK(&plds[wid][0], PSTRIDE, 16, kc * 32, lane);
#pragma unroll
            for (int nj = 0; nj < 4; ++nj) {
                v16h vf = frag_colK(vb, SEQ, nj * 16, key0 + kc * 32, lane);
                C[0][nj] = wmma_f16(pa0, vf, C[0][nj]);
                C[1][nj] = wmma_f16(pa1, vf, C[1][nj]);
            }
        }
    }

    // ---- normalize + store context f16 [BS, 1024] ----
    const int h_ = bh & 15, b_ = bh >> 4;
#pragma unroll
    for (int ti = 0; ti < 2; ++ti) {
#pragma unroll
        for (int nj = 0; nj < 4; ++nj) {
#pragma unroll
            for (int r = 0; r < 8; ++r) {
                const float inv = 1.0f / fmaxf(lrow[ti][r], 1e-30f);
                const int s   = q0 + ti * 16 + hi * 8 + r;
                const int col = h_ * HD + nj * 16 + l;
                ctx[((long)(b_ * SEQ + s)) * EMBED + col] = (h16)(C[ti][nj][r] * inv);
            }
        }
    }
}

// ---------------------------------------------------------------------------
// GEMM2: ctx[8192,1024] @ WprojT[1024,1024] + bias -> out fp32
// ---------------------------------------------------------------------------
__global__ __launch_bounds__(256) void gemm_out(
    const h16* __restrict__ A, const h16* __restrict__ Wt,
    const float* __restrict__ bias, float* __restrict__ out) {
    const int lane = threadIdx.x & 31;
    const int wid  = threadIdx.x >> 5;
    const int row0 = blockIdx.x * 64 + (wid & 1) * 32;
    const int n0   = blockIdx.y * 256 + (wid >> 1) * 64;

    v8f C[2][4];
#pragma unroll
    for (int i = 0; i < 2; ++i)
#pragma unroll
        for (int j = 0; j < 4; ++j)
#pragma unroll
            for (int r = 0; r < 8; ++r) C[i][j][r] = 0.0f;

    for (int k0 = 0; k0 < EMBED; k0 += 32) {
        v16h a0 = frag_rowK(A, EMBED, row0,      k0, lane);
        v16h a1 = frag_rowK(A, EMBED, row0 + 16, k0, lane);
#pragma unroll
        for (int tj = 0; tj < 4; ++tj) {
            v16h bf = frag_colK(Wt, EMBED, n0 + tj * 16, k0, lane);
            C[0][tj] = wmma_f16(a0, bf, C[0][tj]);
            C[1][tj] = wmma_f16(a1, bf, C[1][tj]);
        }
    }

    const int l = lane & 15, hi = lane >> 4;
#pragma unroll
    for (int tj = 0; tj < 4; ++tj) {
        const int n    = n0 + tj * 16 + l;
        const float bv = bias[n];
#pragma unroll
        for (int ti = 0; ti < 2; ++ti) {
#pragma unroll
            for (int r = 0; r < 8; ++r) {
                const int m = row0 + ti * 16 + hi * 8 + r;
                out[(long)m * EMBED + n] = C[ti][tj][r] + bv;
            }
        }
    }
}

// ---------------------------------------------------------------------------
// Host launch
// ---------------------------------------------------------------------------
extern "C" void kernel_launch(void* const* d_in, const int* in_sizes, int n_in,
                              void* d_out, int out_size, void* d_ws, size_t ws_size,
                              hipStream_t stream) {
    const float* x      = (const float*)d_in[0];   // [4,2048,1024]
    const float* w_qkv  = (const float*)d_in[1];   // [1024,3072]
    const float* b_qkv  = (const float*)d_in[2];   // [3072]
    const float* w_proj = (const float*)d_in[3];   // [1024,1024]
    const float* b_proj = (const float*)d_in[4];   // [1024]
    float* out = (float*)d_out;

    char* ws = (char*)d_ws;
    size_t off = 0;
    h16* xh     = (h16*)(ws + off); off += (size_t)BS * EMBED * 2;
    h16* wqkvT  = (h16*)(ws + off); off += (size_t)NQKV * EMBED * 2;
    h16* wprojT = (h16*)(ws + off); off += (size_t)EMBED * EMBED * 2;
    h16* qh     = (h16*)(ws + off); off += (size_t)BS * EMBED * 2;
    h16* kh     = (h16*)(ws + off); off += (size_t)BS * EMBED * 2;
    h16* vth    = (h16*)(ws + off); off += (size_t)BS * EMBED * 2;
    h16* ctx    = (h16*)(ws + off); off += (size_t)BS * EMBED * 2;

    {
        int n = BS * EMBED;
        cvt_f32_to_f16<<<(n + 255) / 256, 256, 0, stream>>>(x, xh, n);
    }
    {
        int n = EMBED * NQKV;
        cvt_transpose_f16<<<(n + 255) / 256, 256, 0, stream>>>(w_qkv, wqkvT, EMBED, NQKV);
    }
    {
        int n = EMBED * EMBED;
        cvt_transpose_f16<<<(n + 255) / 256, 256, 0, stream>>>(w_proj, wprojT, EMBED, EMBED);
    }

    gemm_qkv<<<dim3(BS / 64, NQKV / 256), 256, 0, stream>>>(xh, wqkvT, b_qkv, qh, kh, vth);
    attn_flash<<<dim3(512), 256, 0, stream>>>(qh, kh, vth, ctx);
    gemm_out<<<dim3(BS / 64, EMBED / 256), 256, 0, stream>>>(ctx, wprojT, b_proj, out);
}
